// BivectorRotarySelfAttention_884763263730
// MI455X (gfx1250) — matrix operand order repaired and checked by reading, not compiled, not run on hardware
//
#include <hip/hip_runtime.h>
#include <hip/hip_bf16.h>
#include <math.h>

// ---------------- problem constants ----------------
#define Bsz   4
#define Lseq  1024
#define Dm    2048
#define Hn    16
#define HKVn  4
#define HDd   128
#define HD2d  64
#define NKV   (HKVn * HDd)      // 512
#define MROWS (Bsz * Lseq)      // 4096

typedef __bf16 bf16_t;
typedef __attribute__((ext_vector_type(16))) __bf16        v16bf;
typedef __attribute__((ext_vector_type(8)))  __bf16        v8bf;
typedef __attribute__((ext_vector_type(8)))  float         v8f;
typedef __attribute__((ext_vector_type(4)))  unsigned int  u32x4;
typedef __attribute__((ext_vector_type(4)))  int           i32x4;
typedef __attribute__((ext_vector_type(8)))  int           i32x8;

// address-space qualified pointee types for the async builtin
typedef __attribute__((address_space(1))) i32x4 g_i32x4;   // global
typedef __attribute__((address_space(3))) i32x4 l_i32x4;   // LDS
typedef __attribute__((address_space(3))) void  lds_void_t;

// ---- CDNA5 async / TDM feature detection (compile-safe) ----
#define HAS_ASYNC_LDS __has_builtin(__builtin_amdgcn_global_load_async_to_lds_b128)
#define HAS_TDM       __has_builtin(__builtin_amdgcn_tensor_load_to_lds)

__device__ __forceinline__ void async_cp16(void* lds_dst, const void* gsrc) {
#if HAS_ASYNC_LDS
  __builtin_amdgcn_global_load_async_to_lds_b128(
      (g_i32x4*)gsrc, (l_i32x4*)lds_dst, 0, 0);
#else
  *(v8bf*)lds_dst = *(const v8bf*)gsrc;
#endif
}
__device__ __forceinline__ void wait_async() {
#if HAS_ASYNC_LDS
#if __has_builtin(__builtin_amdgcn_s_wait_asynccnt)
  __builtin_amdgcn_s_wait_asynccnt(0);
#else
  asm volatile("s_wait_asynccnt 0" ::: "memory");
#endif
#endif
}
__device__ __forceinline__ unsigned lds_off32(const void* p) {
  return (unsigned)(unsigned long long)(lds_void_t*)(void*)(const_cast<void*>(p));
}

__device__ __forceinline__ v8f vzero8f() {
  v8f v;
#pragma unroll
  for (int i = 0; i < 8; ++i) v[i] = 0.f;
  return v;
}
__device__ __forceinline__ v16bf cat8(v8bf lo, v8bf hi) {
  return __builtin_shufflevector(lo, hi, 0,1,2,3,4,5,6,7,8,9,10,11,12,13,14,15);
}
__device__ __forceinline__ v8f wmma_bf16(v16bf a, v16bf b, v8f c) {
  return __builtin_amdgcn_wmma_f32_16x16x32_bf16(false, a, false, b, (short)0, c, false, false);
}

// ---------------- fp32 -> bf16 conversion ----------------
__global__ __launch_bounds__(256)
void k_f32_to_bf16(const float* __restrict__ in, bf16_t* __restrict__ out, int n) {
  int i = blockIdx.x * 256 + threadIdx.x;
  if (i < n) out[i] = (bf16_t)in[i];
}

// ---------------- generic bf16 GEMM: C(MxN) = A(MxK) @ B(KxN) ----------------
// Block: 256 threads = 8 waves (4 row-waves x 2 col-waves); tile 128x128;
// wave computes 32x64 via 2x4 WMMA 16x16 tiles; K stepped by 32.
template<bool F32OUT>
__global__ __launch_bounds__(256)
void k_gemm_bf16(const bf16_t* __restrict__ A, const bf16_t* __restrict__ Bm,
                 float* __restrict__ Cf, bf16_t* __restrict__ Cb,
                 int M, int N, int K) {
  __shared__ __align__(32) bf16_t As[128][32];
  __shared__ __align__(32) bf16_t Bs[128][32];   // transposed: [n][k]

  const int tid  = threadIdx.x;
  const int lane = tid & 31;
  const int wave = tid >> 5;
  const int wr   = wave >> 1;           // 0..3
  const int wc   = wave & 1;            // 0..1
  const int half = lane >> 4;           // 0/1
  const int l16  = lane & 15;
  const int m0   = blockIdx.x * 128;
  const int n0   = blockIdx.y * 128;
  const int aoff = half * 16;           // A frag: lanes0-15 K0..15, lanes16-31 K16..31
  const int boff = half * 16;           // B frag: lanes0-15 K0..15, lanes16-31 K16..31

  v8f acc[2][4];
#pragma unroll
  for (int i = 0; i < 2; ++i)
#pragma unroll
    for (int j = 0; j < 4; ++j) acc[i][j] = vzero8f();

  for (int k0 = 0; k0 < K; k0 += 32) {
    // ---- stage A tile (128x32): 512 x 16B chunks, async global->LDS ----
#pragma unroll
    for (int c = 0; c < 2; ++c) {
      int chunk = tid + c * 256;
      int row   = chunk >> 2;
      int col8  = (chunk & 3) * 8;
      async_cp16(&As[row][col8], A + (size_t)(m0 + row) * K + k0 + col8);
    }
    // ---- stage B tile transposed (32x128 -> Bs[n][k]) ----
#pragma unroll
    for (int c = 0; c < 2; ++c) {
      int chunk = tid + c * 256;
      int kr    = chunk >> 4;
      int nb    = (chunk & 15) * 8;
      v8bf vsrc = *(const v8bf*)(Bm + (size_t)(k0 + kr) * N + n0 + nb);
#pragma unroll
      for (int e = 0; e < 8; ++e) Bs[nb + e][kr] = vsrc[e];
    }
    if (k0 + 32 < K) {  // speculative prefetch of next K-slice -> global_prefetch_b8
      __builtin_prefetch(A + (size_t)(m0 + (tid >> 1)) * K + k0 + 32, 0, 0);
      __builtin_prefetch(Bm + (size_t)(k0 + 32 + (tid & 31)) * N + n0, 0, 0);
    }
    wait_async();
    __syncthreads();

    v16bf afrag[2];
#pragma unroll
    for (int i = 0; i < 2; ++i) {
      int row = wr * 32 + i * 16 + l16;
      v8bf lo = *(const v8bf*)&As[row][aoff];
      v8bf hi = *(const v8bf*)&As[row][aoff + 8];
      afrag[i] = cat8(lo, hi);
    }
    v16bf bfrag[4];
#pragma unroll
    for (int j = 0; j < 4; ++j) {
      int col = wc * 64 + j * 16 + l16;
      bfrag[j] = *(const v16bf*)&Bs[col][boff];
    }
#pragma unroll
    for (int i = 0; i < 2; ++i)
#pragma unroll
      for (int j = 0; j < 4; ++j)
        acc[i][j] = wmma_bf16(afrag[i], bfrag[j], acc[i][j]);
    __syncthreads();
  }

  // C layout: vgpr r holds (m = r + half*8, n = l16)
#pragma unroll
  for (int i = 0; i < 2; ++i)
#pragma unroll
    for (int j = 0; j < 4; ++j)
#pragma unroll
      for (int r = 0; r < 8; ++r) {
        int row = m0 + wr * 32 + i * 16 + r + half * 8;
        int col = n0 + wc * 64 + j * 16 + l16;
        if (F32OUT) Cf[(size_t)row * N + col] = acc[i][j][r];
        else        Cb[(size_t)row * N + col] = (bf16_t)acc[i][j][r];
      }
}

// ---------------- RoPE (in place, bf16) ----------------
__global__ __launch_bounds__(256)
void k_rope(bf16_t* __restrict__ x, const float* __restrict__ cosb,
            const float* __restrict__ sinb, int nh, int total) {
  int i = blockIdx.x * 256 + threadIdx.x;
  if (i >= total) return;
  int d  = i & 31;
  int t  = i >> 5;
  int hs = t & 1; t >>= 1;
  int h  = t % nh; t /= nh;
  int l  = t % Lseq;
  int b  = t / Lseq;
  size_t base = ((size_t)(b * Lseq + l) * nh + h) * HDd + (size_t)hs * HD2d;
  float a0 = (float)x[base + d];
  float a1 = (float)x[base + d + 32];
  size_t cb = ((size_t)l * Hn + h) * HD2d;   // cos/sin: (1,L,H,64); k uses heads < HKV
  float c0 = cosb[cb + d],      s0 = sinb[cb + d];
  float c1 = cosb[cb + d + 32], s1 = sinb[cb + d + 32];
  x[base + d]      = (bf16_t)(a0 * c0 - a1 * s0);
  x[base + d + 32] = (bf16_t)(a1 * c1 + a0 * s1);
}

// ---------------- fused bivector flash attention ----------------
// grid: (B*H, L/64); block: 128 threads = 4 waves; wave owns 16 query rows.
__global__ __launch_bounds__(128)
void k_attn(const bf16_t* __restrict__ Q, const bf16_t* __restrict__ Kr,
            const bf16_t* __restrict__ V, bf16_t* __restrict__ O,
            const float* __restrict__ qp, const float* __restrict__ ls,
            const int* __restrict__ mask) {
  __shared__ __align__(32) bf16_t Qs[64][128];
  __shared__ __align__(32) bf16_t Ks[64][128];
  __shared__ __align__(32) bf16_t Vt[128][64];   // transposed V tile: [d][key]
  __shared__ __align__(32) bf16_t Ps[4][16][64]; // per-wave P tile

  const int tid  = threadIdx.x;
  const int lane = tid & 31;
  const int wave = tid >> 5;
  const int half = lane >> 4;
  const int l16  = lane & 15;
  const int bh   = blockIdx.x;
  const int b    = bh / Hn;
  const int h    = bh % Hn;
  const int hk   = h / (Hn / HKVn);
  const int qt   = blockIdx.y;            // query tile (64 rows)
  const int aoff = half * 16;
  const int boff = half * 16;

  const float coef = 2.f * tanhf(qp[h]);
  const float scal = expf(ls[h]) / (float)HDd;

  // ---- stage Q tile (64x128) async ----
#pragma unroll
  for (int c = 0; c < 8; ++c) {
    int chunk = tid + c * 128;
    int row   = chunk >> 4;
    int col8  = (chunk & 15) * 8;
    int ql    = qt * 64 + row;
    async_cp16(&Qs[row][col8],
               Q + ((size_t)(b * Lseq + ql) * Hn + h) * HDd + col8);
  }
  wait_async();
  __syncthreads();

  // Q frags (invariant over key tiles): [hh = q0/q1][kstep]
  const int mrow = wave * 16 + l16;
  v16bf qf[2][2];
#pragma unroll
  for (int hh = 0; hh < 2; ++hh)
#pragma unroll
    for (int st = 0; st < 2; ++st) {
      int kb = hh * 64 + st * 32;
      v8bf lo = *(const v8bf*)&Qs[mrow][kb + aoff];
      v8bf hi = *(const v8bf*)&Qs[mrow][kb + aoff + 8];
      qf[hh][st] = cat8(lo, hi);
    }

  v8f oacc[8];
#pragma unroll
  for (int d = 0; d < 8; ++d) oacc[d] = vzero8f();
  float mrun[8], lrun[8];
#pragma unroll
  for (int r = 0; r < 8; ++r) { mrun[r] = -1e30f; lrun[r] = 0.f; }

  for (int kt = 0; kt <= qt; ++kt) {
    __syncthreads();   // previous iteration's reads of Ks/Vt are done

#if HAS_TDM
    // ---- K tile (64x128, row stride NKV) via Tensor Data Mover, wave 0 only ----
    if (wave == 0) {
      unsigned ldsa = lds_off32(&Ks[0][0]);
      unsigned long long ga = (unsigned long long)
          (Kr + ((size_t)(b * Lseq + kt * 64) * HKVn + hk) * HDd);
      u32x4 g0;
      g0[0] = 1u;                                           // count=1, user mode
      g0[1] = ldsa;                                         // lds_addr
      g0[2] = (unsigned)ga;                                 // global_addr[31:0]
      g0[3] = (unsigned)((ga >> 32) & 0x1FFFFFFu) | (2u << 30); // addr[56:32] | type=2
      i32x8 g1;
      g1[0] = (1 << 16);          // wg_mask=0, data_size=1 (2 bytes)
      g1[1] = (HDd << 16);        // tensor_dim0[15:0] = 128
      g1[2] = (64 << 16);         // tensor_dim0 hi=0 | tensor_dim1[15:0] = 64
      g1[3] = (HDd << 16);        // tensor_dim1 hi=0 | tile_dim0 = 128
      g1[4] = 64;                 // tile_dim1 = 64, tile_dim2 = 0
      g1[5] = NKV;                // tensor_dim0_stride = 512 elements
      g1[6] = 0;
      g1[7] = 0;
      i32x4 gz;
      gz[0] = 0; gz[1] = 0; gz[2] = 0; gz[3] = 0;
#if __has_include(<hip/amd_detail/amd_gfx1250_TDM.h>)
      i32x8 gz8;
#pragma unroll
      for (int e = 0; e < 8; ++e) gz8[e] = 0;
      __builtin_amdgcn_tensor_load_to_lds(g0, g1, gz, gz, gz8, 0);   // therock 6-arg form
#else
      __builtin_amdgcn_tensor_load_to_lds(g0, g1, gz, gz, 0);        // ROCm 7.2 5-arg form
#endif
      __builtin_amdgcn_s_wait_tensorcnt(0);
    }
#endif

    // ---- V tile transposed (manual scatter); K tile async if no TDM ----
#pragma unroll
    for (int c = 0; c < 8; ++c) {
      int chunk = tid + c * 128;
      int row   = chunk >> 4;
      int col8  = (chunk & 15) * 8;
      int kl    = kt * 64 + row;
      size_t src = ((size_t)(b * Lseq + kl) * HKVn + hk) * HDd + col8;
#if !HAS_TDM
      async_cp16(&Ks[row][col8], Kr + src);
#endif
      v8bf vv = *(const v8bf*)(V + src);
#pragma unroll
      for (int e = 0; e < 8; ++e) Vt[col8 + e][row] = vv[e];
    }
    wait_async();
    __syncthreads();

    // ---- scores: 4 col-subtiles of 16 keys ----
    v8f Sv[4];
#pragma unroll
    for (int sub = 0; sub < 4; ++sub) {
      v8f s00 = vzero8f(), s11 = vzero8f(), s01 = vzero8f(), s10 = vzero8f();
      int kcol = sub * 16 + l16;
#pragma unroll
      for (int st = 0; st < 2; ++st) {
        v16bf k0f = *(const v16bf*)&Ks[kcol][st * 32 + boff];
        v16bf k1f = *(const v16bf*)&Ks[kcol][64 + st * 32 + boff];
        s00 = wmma_bf16(qf[0][st], k0f, s00);
        s11 = wmma_bf16(qf[1][st], k1f, s11);
        s01 = wmma_bf16(qf[0][st], k1f, s01);
        s10 = wmma_bf16(qf[1][st], k0f, s10);
      }
      int kglob = kt * 64 + sub * 16 + l16;
      int mv    = mask[b * Lseq + kglob];
      v8f sv;
#pragma unroll
      for (int r = 0; r < 8; ++r) {
        float raw = s00[r] * s11[r] - coef * (s01[r] * s10[r]);
        raw *= scal;
        int qglob = qt * 64 + wave * 16 + r + half * 8;
        if (kglob > qglob || mv == 0) raw = -1e9f;
        sv[r] = raw;
      }
      Sv[sub] = sv;
    }

    // ---- online softmax (row m = r + half*8 lives across a 16-lane half) ----
#pragma unroll
    for (int r = 0; r < 8; ++r) {
      float tm = fmaxf(fmaxf(Sv[0][r], Sv[1][r]), fmaxf(Sv[2][r], Sv[3][r]));
#pragma unroll
      for (int x = 8; x >= 1; x >>= 1) tm = fmaxf(tm, __shfl_xor(tm, x, 32));
      float mnew = fmaxf(mrun[r], tm);
      float corr = __expf(mrun[r] - mnew);
      float psum = 0.f;
#pragma unroll
      for (int sub = 0; sub < 4; ++sub) {
        float p = __expf(Sv[sub][r] - mnew);
        Sv[sub][r] = p;
        psum += p;
      }
#pragma unroll
      for (int x = 8; x >= 1; x >>= 1) psum += __shfl_xor(psum, x, 32);
      lrun[r] = lrun[r] * corr + psum;
      mrun[r] = mnew;
#pragma unroll
      for (int ds = 0; ds < 8; ++ds) oacc[ds][r] *= corr;
    }

    // ---- P -> per-wave LDS, reshape to A-frags ----
#pragma unroll
    for (int sub = 0; sub < 4; ++sub)
#pragma unroll
      for (int r = 0; r < 8; ++r)
        Ps[wave][r + half * 8][sub * 16 + l16] = (bf16_t)Sv[sub][r];
    asm volatile("s_wait_dscnt 0" ::: "memory");   // own-wave LDS RAW

    // ---- O += P @ V ----
#pragma unroll
    for (int st = 0; st < 2; ++st) {
      v8bf lo = *(const v8bf*)&Ps[wave][l16][st * 32 + aoff];
      v8bf hi = *(const v8bf*)&Ps[wave][l16][st * 32 + aoff + 8];
      v16bf pf = cat8(lo, hi);
#pragma unroll
      for (int ds = 0; ds < 8; ++ds) {
        v16bf vf = *(const v16bf*)&Vt[ds * 16 + l16][st * 32 + boff];
        oacc[ds] = wmma_bf16(pf, vf, oacc[ds]);
      }
    }
  }

  // ---- normalize + store (B,L,H,HD) == (B,L,D) bf16 ----
#pragma unroll
  for (int ds = 0; ds < 8; ++ds)
#pragma unroll
    for (int r = 0; r < 8; ++r) {
      int qglob = qt * 64 + wave * 16 + r + half * 8;
      float o = oacc[ds][r] / lrun[r];
      O[(size_t)(b * Lseq + qglob) * Dm + h * HDd + ds * 16 + l16] = (bf16_t)o;
    }
}

// ---------------- host launcher ----------------
extern "C" void kernel_launch(void* const* d_in, const int* in_sizes, int n_in,
                              void* d_out, int out_size, void* d_ws, size_t ws_size,
                              hipStream_t stream) {
  const float* x    = (const float*)d_in[0];
  const float* Wq   = (const float*)d_in[1];
  const float* Wk   = (const float*)d_in[2];
  const float* Wv   = (const float*)d_in[3];
  const float* Wo   = (const float*)d_in[4];
  const float* qp   = (const float*)d_in[5];
  const float* lsc  = (const float*)d_in[6];
  const float* cosb = (const float*)d_in[7];
  const float* sinb = (const float*)d_in[8];
  const int*   msk  = (const int*)d_in[9];

  char* ws = (char*)d_ws;
  bf16_t* xb  = (bf16_t*)(ws + 0);            // 8M el
  bf16_t* wqb = (bf16_t*)(ws + (16u << 20));  // 4M el
  bf16_t* wkb = (bf16_t*)(ws + (24u << 20));  // 1M el
  bf16_t* wvb = (bf16_t*)(ws + (26u << 20));  // 1M el
  bf16_t* wob = (bf16_t*)(ws + (28u << 20));  // 4M el
  bf16_t* qb  = (bf16_t*)(ws + (36u << 20));  // 8M el
  bf16_t* kb  = (bf16_t*)(ws + (52u << 20));  // 2M el
  bf16_t* vb  = (bf16_t*)(ws + (56u << 20));  // 2M el
  bf16_t* ab  = (bf16_t*)(ws + (60u << 20));  // 8M el

  const int nX  = MROWS * Dm;          // 8388608
  const int nWq = Dm * Dm;             // 4194304
  const int nWk = Dm * NKV;            // 1048576

  k_f32_to_bf16<<<(nX + 255) / 256, 256, 0, stream>>>(x, xb, nX);
  k_f32_to_bf16<<<(nWq + 255) / 256, 256, 0, stream>>>(Wq, wqb, nWq);
  k_f32_to_bf16<<<(nWk + 255) / 256, 256, 0, stream>>>(Wk, wkb, nWk);
  k_f32_to_bf16<<<(nWk + 255) / 256, 256, 0, stream>>>(Wv, wvb, nWk);
  k_f32_to_bf16<<<(nWq + 255) / 256, 256, 0, stream>>>(Wo, wob, nWq);

  // projections
  k_gemm_bf16<false><<<dim3(MROWS / 128, Dm / 128), 256, 0, stream>>>(
      xb, wqb, nullptr, qb, MROWS, Dm, Dm);
  k_gemm_bf16<false><<<dim3(MROWS / 128, NKV / 128), 256, 0, stream>>>(
      xb, wkb, nullptr, kb, MROWS, NKV, Dm);
  k_gemm_bf16<false><<<dim3(MROWS / 128, NKV / 128), 256, 0, stream>>>(
      xb, wvb, nullptr, vb, MROWS, NKV, Dm);

  // RoPE on q (16 heads) and k (4 heads)
  const int totQ = Bsz * Lseq * Hn * 64;
  const int totK = Bsz * Lseq * HKVn * 64;
  k_rope<<<(totQ + 255) / 256, 256, 0, stream>>>(qb, cosb, sinb, Hn, totQ);
  k_rope<<<(totK + 255) / 256, 256, 0, stream>>>(kb, cosb, sinb, HKVn, totK);

  // fused attention
  k_attn<<<dim3(Bsz * Hn, Lseq / 64), 128, 0, stream>>>(qb, kb, vb, ab, qp, lsc, msk);

  // output projection -> fp32 d_out
  k_gemm_bf16<true><<<dim3(MROWS / 128, Dm / 128), 256, 0, stream>>>(
      ab, wob, (float*)d_out, nullptr, MROWS, Dm, Dm);
}
